// VectorQuantizer_34677565948518
// MI455X (gfx1250) — matrix operand order repaired and checked by reading, not compiled, not run on hardware
//
#include <hip/hip_runtime.h>
#include <hip/hip_bf16.h>

// ---------------------------------------------------------------------------
// VQ codebook lookup for MI455X (gfx1250, wave32, WMMA bf16 16x16x32)
//   inputs     : [64,64,64,64] f32   (262144 pixels x 64 dims)
//   embeddings : [512,64]      f32
//   out        : [262144,64]   f32   (selected embedding per pixel)
// argmin(||x-e||^2) == argmin(||e||^2 - 2 x.e)  -> pure GEMM + epilogue.
// ---------------------------------------------------------------------------

#define NPIX         (64 * 64 * 64)
#define KCODES       512
#define DIMS         64
#define BLOCK        256          // 8 waves
#define WAVES        8
#define PASSES       2
#define PIX_PER_PASS 32           // per wave: two 16x16 WMMA tiles
#define PIX_PER_WG   (WAVES * PASSES * PIX_PER_PASS)  // 512
#define GRID         (NPIX / PIX_PER_WG)              // 512

typedef __attribute__((ext_vector_type(16))) __bf16 v16bf;
typedef __attribute__((ext_vector_type(8)))  float  v8f;

struct BPair { uint4 lo, hi; };  // 32B -> bit_cast to v16bf (no union/SROA issues)

// ---------------------------------------------------------------------------
// Prep: fp32 embeddings -> packed bf16 in d_ws, plus l2e = ||e||^2.
// ---------------------------------------------------------------------------
__global__ void vq_prep(const float* __restrict__ emb,
                        unsigned int* __restrict__ wsbf,
                        float* __restrict__ l2e) {
  int code = blockIdx.x * blockDim.x + threadIdx.x;
  if (code >= KCODES) return;
  const float* r = emb + code * DIMS;
  unsigned int* dst = wsbf + code * (DIMS / 2);
  float s = 0.0f;
#pragma unroll
  for (int d2 = 0; d2 < DIMS / 2; ++d2) {
    float v0 = r[2 * d2 + 0];
    float v1 = r[2 * d2 + 1];
    s = fmaf(v0, v0, fmaf(v1, v1, s));
    unsigned short b0 = __builtin_bit_cast(unsigned short, (__bf16)v0);
    unsigned short b1 = __builtin_bit_cast(unsigned short, (__bf16)v1);
    dst[d2] = (unsigned int)b0 | ((unsigned int)b1 << 16);
  }
  l2e[code] = s;
}

// A-fragment (contiguous k = 16h + s, same order as B): lane half h, chunk c needs fp32 dims
// [32c+16h, +16)  -> one 16-float run, 4x float4 loads.
__device__ __forceinline__ v16bf load_a_frag(const float* __restrict__ rowp,
                                             int c, int h) {
  const float4* p0 = reinterpret_cast<const float4*>(rowp + 32 * c + 16 * h);
  const float4* p1 = reinterpret_cast<const float4*>(rowp + 32 * c + 16 * h + 8);
  float4 x0 = p0[0], x1 = p0[1];
  float4 x2 = p1[0], x3 = p1[1];
  v16bf a;
  a[0]  = (__bf16)x0.x;  a[1]  = (__bf16)x0.y;
  a[2]  = (__bf16)x0.z;  a[3]  = (__bf16)x0.w;
  a[4]  = (__bf16)x1.x;  a[5]  = (__bf16)x1.y;
  a[6]  = (__bf16)x1.z;  a[7]  = (__bf16)x1.w;
  a[8]  = (__bf16)x2.x;  a[9]  = (__bf16)x2.y;
  a[10] = (__bf16)x2.z;  a[11] = (__bf16)x2.w;
  a[12] = (__bf16)x3.x;  a[13] = (__bf16)x3.y;
  a[14] = (__bf16)x3.z;  a[15] = (__bf16)x3.w;
  return a;
}

// Monotone float -> uint (order-preserving for all finite values).
__device__ __forceinline__ unsigned monof(float f) {
  unsigned u = __builtin_bit_cast(unsigned, f);
  return u ^ ((unsigned)((int)u >> 31) | 0x80000000u);
}

// ---------------------------------------------------------------------------
// Main kernel.
// ---------------------------------------------------------------------------
__global__ __launch_bounds__(BLOCK)
__attribute__((amdgpu_waves_per_eu(1)))
void vq_main(
    const float* __restrict__ in, const float* __restrict__ embf,
    const unsigned int* __restrict__ wsbf, const float* __restrict__ l2e,
    float* __restrict__ out) {
  // 64KB swizzled bf16 codebook: uint4 slot = code*8 + (q ^ ((code>>1)&7)).
  __shared__ uint4 semb[KCODES * 8];
  {
    const uint4* src = reinterpret_cast<const uint4*>(wsbf);
    for (int i = threadIdx.x; i < KCODES * 8; i += BLOCK) {
      int code = i >> 3, q = i & 7;
      semb[(code << 3) | (q ^ ((code >> 1) & 7))] = src[i];
    }
  }
  __syncthreads();

  const int lane = threadIdx.x & 31;
  const int wv   = threadIdx.x >> 5;
  const int n    = lane & 15;    // column / code-in-block / pixel row
  const int h    = lane >> 4;    // half: selects K-subgroup in A/B layouts
  const int sw   = n >> 1;       // LDS bank swizzle (== (code>>1)&7 here)

  // Per-lane swizzled base pointers; cb contributes a constant ds-offset.
  const uint4* sp0 = &semb[(n << 3) | ((2 * h + 0) ^ sw)];
  const uint4* sp1 = &semb[(n << 3) | ((2 * h + 1) ^ sw)];
  const uint4* sp2 = &semb[(n << 3) | ((4 + 2 * h + 0) ^ sw)];
  const uint4* sp3 = &semb[(n << 3) | ((4 + 2 * h + 1) ^ sw)];

  const uint4* embq = reinterpret_cast<const uint4*>(embf);
  uint4* outq = reinterpret_cast<uint4*>(out);

  const int wgBase = blockIdx.x * PIX_PER_WG;

  for (int pass = 0; pass < PASSES; ++pass) {
    const int pbase = wgBase + (wv * PASSES + pass) * PIX_PER_PASS;

    const float* row0 = in + (size_t)(pbase + n) * DIMS;
    const float* row1 = in + (size_t)(pbase + 16 + n) * DIMS;
    v16bf a00 = load_a_frag(row0, 0, h);
    v16bf a01 = load_a_frag(row0, 1, h);
    v16bf a10 = load_a_frag(row1, 0, h);
    v16bf a11 = load_a_frag(row1, 1, h);

    if (pass + 1 < PASSES) {  // warm the path for next pass's A tiles
      __builtin_prefetch(row0 + PIX_PER_PASS * DIMS, 0, 0);
      __builtin_prefetch(row0 + PIX_PER_PASS * DIMS + 32, 0, 0);
      __builtin_prefetch(row1 + PIX_PER_PASS * DIMS, 0, 0);
      __builtin_prefetch(row1 + PIX_PER_PASS * DIMS + 32, 0, 0);
    }

    // Fence: keep the main-loop ds_loads / l2e loads from being hoisted into
    // the A-staging region (caps peak VGPR demand -> no spills).
    __builtin_amdgcn_sched_barrier(0);

    float minv0[8], minv1[8];
    int   mini0[8], mini1[8];
#pragma unroll
    for (int r = 0; r < 8; ++r) {
      minv0[r] = 3.0e38f; minv1[r] = 3.0e38f;
      mini0[r] = 0;       mini1[r] = 0;
    }

    // Two unrolled halves of 16 code-blocks each; only 16 l2e values live at
    // a time (keeps peak VGPR pressure down).
#pragma unroll
    for (int hb = 0; hb < 2; ++hb) {
      float l2h[16];
#pragma unroll
      for (int j = 0; j < 16; ++j) l2h[j] = l2e[((hb * 16 + j) << 4) | n];

#pragma unroll
      for (int j = 0; j < 16; ++j) {
        const int cb = hb * 16 + j;
        const int code = (cb << 4) | n;
        BPair p0, p1;
        p0.lo = sp0[cb << 7];   // chunk0: dims 16h..16h+15
        p0.hi = sp1[cb << 7];
        p1.lo = sp2[cb << 7];   // chunk1: dims 32+16h..
        p1.hi = sp3[cb << 7];
        const v16bf b0 = __builtin_bit_cast(v16bf, p0);
        const v16bf b1 = __builtin_bit_cast(v16bf, p1);
        const float l2 = l2h[j];

        v8f zero = {};
        v8f acc0 = __builtin_amdgcn_wmma_f32_16x16x32_bf16(
            false, a00, false, b0, (short)0, zero, false, false);
        acc0 = __builtin_amdgcn_wmma_f32_16x16x32_bf16(
            false, a01, false, b1, (short)0, acc0, false, false);
        v8f acc1 = __builtin_amdgcn_wmma_f32_16x16x32_bf16(
            false, a10, false, b0, (short)0, zero, false, false);
        acc1 = __builtin_amdgcn_wmma_f32_16x16x32_bf16(
            false, a11, false, b1, (short)0, acc1, false, false);

#pragma unroll
        for (int r = 0; r < 8; ++r) {
          float d0 = fmaf(acc0[r], -2.0f, l2);
          bool t0 = d0 < minv0[r];
          minv0[r] = t0 ? d0 : minv0[r];
          mini0[r] = t0 ? code : mini0[r];
          float d1 = fmaf(acc1[r], -2.0f, l2);
          bool t1 = d1 < minv1[r];
          minv1[r] = t1 ? d1 : minv1[r];
          mini1[r] = t1 ? code : mini1[r];
        }
      }
    }

    // Cross-lane argmin over the 16 lanes of each half via packed sortable
    // keys: [31:9]=monotone(distance) msbs, [8:0]=code. min_u32 reduction
    // gives lowest-code-on-tie, matching jnp.argmin.
    unsigned key0[8], key1[8];
#pragma unroll
    for (int r = 0; r < 8; ++r) {
      key0[r] = (monof(minv0[r]) & 0xFFFFFE00u) | (unsigned)mini0[r];
      key1[r] = (monof(minv1[r]) & 0xFFFFFE00u) | (unsigned)mini1[r];
    }
#pragma unroll
    for (int off = 1; off < 16; off <<= 1) {
#pragma unroll
      for (int r = 0; r < 8; ++r) {
        unsigned o0 = __shfl_xor(key0[r], off, 32);
        key0[r] = key0[r] < o0 ? key0[r] : o0;
        unsigned o1 = __shfl_xor(key1[r], off, 32);
        key1[r] = key1[r] < o1 ? key1[r] : o1;
      }
    }

    // Gather: lane (n,h) writes float4 chunk c4=n of pixels {tile*16+8h+r}.
    // Codes are uniform across the 16 lanes of a half -> coalesced 256B stores.
#pragma unroll
    for (int t = 0; t < 16; ++t) {
      const int r = t & 7, tile = t >> 3;
      const int code = (int)((tile ? key1[r] : key0[r]) & 511u);
      const int p = tile * 16 + 8 * h + r;
      outq[(size_t)(pbase + p) * 16 + n] = embq[(size_t)code * 16 + n];
    }
  }
}

// ---------------------------------------------------------------------------
extern "C" void kernel_launch(void* const* d_in, const int* in_sizes, int n_in,
                              void* d_out, int out_size, void* d_ws,
                              size_t ws_size, hipStream_t stream) {
  (void)in_sizes; (void)n_in; (void)out_size; (void)ws_size;
  const float* in  = (const float*)d_in[0];   // [64,64,64,64] f32
  const float* emb = (const float*)d_in[1];   // [512,64]      f32
  unsigned int* wsbf = (unsigned int*)d_ws;                    // 64 KB bf16 codebook
  float* l2e = (float*)((char*)d_ws + KCODES * DIMS * 2);      // 2 KB ||e||^2
  float* out = (float*)d_out;

  vq_prep<<<2, 256, 0, stream>>>(emb, wsbf, l2e);
  vq_main<<<GRID, BLOCK, 0, stream>>>(in, emb, wsbf, l2e, out);
}
